// TimeRNNBlock_33878702030939
// MI455X (gfx1250) — compile-verified, matrix operand order checked by reading, not run on hardware
//
#include <hip/hip_runtime.h>
#include <math.h>

// ---------------- CDNA5 WMMA types ----------------
typedef __attribute__((ext_vector_type(16))) __bf16    v16bf;
typedef __attribute__((ext_vector_type(8)))  float     v8f;
typedef __attribute__((ext_vector_type(4)))  unsigned  u32x4;

union ABF {
    u32x4        q[2];   // two 16B chunks (global_load_b128)
    unsigned int u[8];   // eight bf16 pairs (ds_load_b32)
    v16bf        v;      // WMMA operand
};

// ---------------- problem constants ----------------
#define BB   4
#define CC   256
#define TT   512
#define FF   64
#define HH   256
#define OO   256
#define GG   768              // 3*H
#define NSEQ (BB*FF)          // 256 sequences
#define MROW (NSEQ*TT)        // 131072 GEMM rows

// ---------------- bf16 helpers ----------------
__device__ __forceinline__ unsigned short f32_to_bf16(float f) {
    unsigned int u = __float_as_uint(f);
    u = u + 0x7FFFu + ((u >> 16) & 1u);          // round-to-nearest-even
    return (unsigned short)(u >> 16);
}
__device__ __forceinline__ float bf16_to_f32(unsigned short h) {
    return __uint_as_float(((unsigned int)h) << 16);
}

// K-pair base inside a 16x32 A-tile / 32x16 B-tile for (vgpr-pair v, lane half)
__device__ __forceinline__ int kpair_base(int v, int khalf) {
    return (v < 4 ? 2 * v : 16 + 2 * (v - 4)) + khalf;
}

// load one packed B-operand tile (512 bf16, lane-major, 32B/lane)
__device__ __forceinline__ void load_btile(ABF& bb,
                                           const unsigned short* __restrict__ wt,
                                           long tile, int lane) {
    const u32x4* bp = (const u32x4*)(wt + tile * 512);
    bb.q[0] = bp[lane * 2];
    bb.q[1] = bp[lane * 2 + 1];
}

// load one A-operand tile from a row-major bf16 row pointer (k-pairs per lane)
__device__ __forceinline__ void load_atile(ABF& a,
                                           const unsigned short* __restrict__ ap,
                                           int khalf) {
#pragma unroll
    for (int v = 0; v < 8; ++v)
        a.u[v] = *(const unsigned int*)(ap + kpair_base(v, khalf));
}

// =====================================================================
// 1) Repack f32 weight matrix W[N x K] into WMMA B-operand tiles (bf16).
//    Tile (nt,kt) = 512 bf16 stored lane-major: elem index = lane*16 + j.
// =====================================================================
__global__ void pack_weights(const float* __restrict__ W,
                             unsigned short* __restrict__ outw,
                             int Ktiles, int Kdim, int total) {
    int idx = blockIdx.x * 256 + threadIdx.x;
    if (idx >= total) return;
    int j    = idx & 15;
    int lane = (idx >> 4) & 31;
    int tile = idx >> 9;
    int nt = tile / Ktiles, kt = tile % Ktiles;
    int v = j >> 1, odd = j & 1;
    int k0 = kpair_base(v, (lane >= 16) ? 8 : 0) + odd;
    int n  = nt * 16 + (lane & 15);
    int k  = kt * 32 + k0;
    outw[idx] = f32_to_bf16(W[n * Kdim + k]);
}

// =====================================================================
// 2) input [B,C,T,F] f32  ->  xb [(b*F+f), t, c] bf16   (LDS tile transpose)
//    grid: (C/32, F/32, B*T), block (32,8)
// =====================================================================
__global__ void transpose_in(const float* __restrict__ in,
                             unsigned short* __restrict__ xb) {
    __shared__ float tile[32][33];
    const int bx = blockIdx.x, by = blockIdx.y, bz = blockIdx.z;
    const int b = bz >> 9, t = bz & 511;
    const int tx = threadIdx.x, ty = threadIdx.y;
#pragma unroll
    for (int k = 0; k < 4; ++k) {
        int c = bx * 32 + ty + k * 8;
        int f = by * 32 + tx;
        tile[ty + k * 8][tx] = in[(((long)b * CC + c) * TT + t) * FF + f];
    }
    __syncthreads();
#pragma unroll
    for (int k = 0; k < 4; ++k) {
        int fl = ty + k * 8;
        int n  = b * FF + by * 32 + fl;
        int c  = bx * 32 + tx;
        xb[((long)n * TT + t) * CC + c] = f32_to_bf16(tile[tx][fl]);
    }
}

// =====================================================================
// 3) gx = xb * W_ih^T + b_ih   -> bf16 [MROW x 768]
//    grid 8192 blocks (one 16-row M-tile), 256 thr = 8 waves x 6 N-tiles.
//    A/B operand registers double-buffered across K-steps.
// =====================================================================
__global__ void __launch_bounds__(256)
gx_gemm(const unsigned short* __restrict__ xb,
        const unsigned short* __restrict__ wt,
        const float* __restrict__ b_ih,
        unsigned short* __restrict__ gx) {
    const int mtile = blockIdx.x;
    const int lane  = threadIdx.x & 31;
    const int wave  = threadIdx.x >> 5;
    const int khalf = (lane >= 16) ? 8 : 0;
    const long rowA = (long)mtile * 16 + (lane & 15);
    const unsigned short* arow = xb + rowA * CC;

    v8f acc[6] = {};
    ABF a[2];
    ABF bb[2][6];

    load_atile(a[0], arow, khalf);
#pragma unroll
    for (int i = 0; i < 6; ++i)
        load_btile(bb[0][i], wt, (long)((wave * 6 + i) * 8), lane);

#pragma unroll
    for (int kt = 0; kt < 8; ++kt) {
        const int cur = kt & 1, nxt = cur ^ 1;
        if (kt < 7) {                       // loads for k+1 fly under k's WMMAs
            load_atile(a[nxt], arow + (kt + 1) * 32, khalf);
#pragma unroll
            for (int i = 0; i < 6; ++i)
                load_btile(bb[nxt][i], wt, (long)((wave * 6 + i) * 8 + kt + 1), lane);
        }
#pragma unroll
        for (int i = 0; i < 6; ++i)
            acc[i] = __builtin_amdgcn_wmma_f32_16x16x32_bf16(
                false, a[cur].v, false, bb[cur][i].v, (short)0, acc[i], false, false);
    }

#pragma unroll
    for (int i = 0; i < 6; ++i) {
        int g = (wave * 6 + i) * 16 + (lane & 15);
        float bias = b_ih[g];
#pragma unroll
        for (int j = 0; j < 8; ++j) {
            long r = (long)mtile * 16 + j + khalf;   // D-layout: M = j (+8 hi lanes)
            gx[r * GG + g] = f32_to_bf16(acc[i][j] + bias);
        }
    }
}

// =====================================================================
// 4) Sequential GRU scan. 16 persistent blocks, each owns 16 sequences.
//    Hidden state lives in LDS (f32 master + bf16 A-layout source).
//    B tiles double-buffered; W_hh stream prefetched once (L2-resident).
// =====================================================================
__global__ void __launch_bounds__(256)
gru_scan(const unsigned short* __restrict__ gx,
         const unsigned short* __restrict__ wt,     // packed W_hh tiles
         const float* __restrict__ b_hh,
         unsigned short* __restrict__ yg) {
    __shared__ float          hs[16 * HH];   // h, f32
    __shared__ unsigned short hb[16 * HH];   // h, bf16 (WMMA A source)
    __shared__ float          gh[16 * GG];   // h*W_hh^T for this step

    const int tid  = threadIdx.x;
    const int lane = tid & 31, wave = tid >> 5;
    const int khalf = (lane >= 16) ? 8 : 0;
    const int nblk = blockIdx.x;

    for (int i = tid; i < 16 * HH; i += 256) { hs[i] = 0.0f; hb[i] = 0; }
    // one-time warm of this wave's W_hh tile region (stays L2/WGP$-hot after)
    for (int tl = wave * 6 * 8; tl < (wave * 6 + 6) * 8; ++tl)
        __builtin_prefetch(wt + (long)tl * 512 + lane * 16, 0, 3);
    __syncthreads();

    for (int t = 0; t < TT; ++t) {
        // ---- gh = h * W_hh^T  (8 waves x 6 N-tiles x 8 K-steps of WMMA) ----
        v8f acc[6] = {};
        ABF a[2];
        ABF bb[2][6];
        load_atile(a[0], hb + (lane & 15) * HH, khalf);
#pragma unroll
        for (int i = 0; i < 6; ++i)
            load_btile(bb[0][i], wt, (long)((wave * 6 + i) * 8), lane);

#pragma unroll
        for (int kt = 0; kt < 8; ++kt) {
            const int cur = kt & 1, nxt = cur ^ 1;
            if (kt < 7) {
                load_atile(a[nxt], hb + (lane & 15) * HH + (kt + 1) * 32, khalf);
#pragma unroll
                for (int i = 0; i < 6; ++i)
                    load_btile(bb[nxt][i], wt, (long)((wave * 6 + i) * 8 + kt + 1), lane);
            }
#pragma unroll
            for (int i = 0; i < 6; ++i)
                acc[i] = __builtin_amdgcn_wmma_f32_16x16x32_bf16(
                    false, a[cur].v, false, bb[cur][i].v, (short)0, acc[i], false, false);
        }

#pragma unroll
        for (int i = 0; i < 6; ++i) {
            int g = (wave * 6 + i) * 16 + (lane & 15);
#pragma unroll
            for (int j = 0; j < 8; ++j)
                gh[(j + khalf) * GG + g] = acc[i][j];
        }
        __syncthreads();

        // ---- fused gate math + state update (torch gate order r,z,n) ----
        for (int idx = tid; idx < 16 * HH; idx += 256) {
            int row = idx >> 8, u = idx & 255;
            long n    = (long)nblk * 16 + row;
            long base = (n * TT + t) * GG;
            float xr = bf16_to_f32(gx[base + u]);
            float xz = bf16_to_f32(gx[base + HH + u]);
            float xn = bf16_to_f32(gx[base + 2 * HH + u]);
            float hr = gh[row * GG + u]          + b_hh[u];
            float hz = gh[row * GG + HH + u]     + b_hh[HH + u];
            float hn = gh[row * GG + 2 * HH + u] + b_hh[2 * HH + u];
            float r  = 1.0f / (1.0f + __expf(-(xr + hr)));
            float z  = 1.0f / (1.0f + __expf(-(xz + hz)));
            float nn = tanhf(xn + r * hn);
            float hnew = (1.0f - z) * nn + z * hs[idx];
            hs[idx] = hnew;
            unsigned short hb16 = f32_to_bf16(hnew);
            hb[idx] = hb16;
            yg[(n * TT + t) * HH + u] = hb16;
        }
        __syncthreads();
    }
}

// =====================================================================
// 5) z = yg * fc_w^T + fc_b + skip  -> d_out (f32), plus per-channel
//    sum / sumsq for BatchNorm via LDS reduction + global atomics.
// =====================================================================
__global__ void __launch_bounds__(256)
conv_skip(const unsigned short* __restrict__ yg,
          const unsigned short* __restrict__ wt,   // packed fc_w
          const float* __restrict__ fc_b,
          const float* __restrict__ skip,
          float* __restrict__ out,
          float* __restrict__ gsum,
          float* __restrict__ gsq) {
    __shared__ float lsum[OO], lsq[OO];
    const int mtile = blockIdx.x;
    const int tid = threadIdx.x, lane = tid & 31, wave = tid >> 5;
    const int khalf = (lane >= 16) ? 8 : 0;
    if (tid < OO) { lsum[tid] = 0.0f; lsq[tid] = 0.0f; }
    __syncthreads();

    const long rowA = (long)mtile * 16 + (lane & 15);
    const unsigned short* arow = yg + rowA * HH;
    const int n0 = (mtile * 16) >> 9;        // sequence id (constant in tile)
    const int t0 = (mtile * 16) & 511;
    const int b = n0 >> 6, f = n0 & 63;

    v8f acc[2] = {};
    ABF a[2];
    ABF bb[2][2];
    load_atile(a[0], arow, khalf);
#pragma unroll
    for (int i = 0; i < 2; ++i)
        load_btile(bb[0][i], wt, (long)((wave * 2 + i) * 8), lane);

#pragma unroll
    for (int kt = 0; kt < 8; ++kt) {
        const int cur = kt & 1, nxt = cur ^ 1;
        if (kt < 7) {
            load_atile(a[nxt], arow + (kt + 1) * 32, khalf);
#pragma unroll
            for (int i = 0; i < 2; ++i)
                load_btile(bb[nxt][i], wt, (long)((wave * 2 + i) * 8 + kt + 1), lane);
        }
#pragma unroll
        for (int i = 0; i < 2; ++i)
            acc[i] = __builtin_amdgcn_wmma_f32_16x16x32_bf16(
                false, a[cur].v, false, bb[cur][i].v, (short)0, acc[i], false, false);
    }

#pragma unroll
    for (int i = 0; i < 2; ++i) {
        int o = (wave * 2 + i) * 16 + (lane & 15);
        float bias = fc_b[o];
        long obase = (((long)b * OO + o) * TT + t0) * FF + f;
        float s = 0.0f, s2 = 0.0f;
#pragma unroll
        for (int j = 0; j < 8; ++j) {
            long oi = obase + (long)(j + khalf) * FF;
            float v = acc[i][j] + bias + skip[oi];
            out[oi] = v;
            s += v; s2 += v * v;
        }
        atomicAdd(&lsum[o], s);
        atomicAdd(&lsq[o], s2);
    }
    __syncthreads();
    if (tid < OO) {
        atomicAdd(&gsum[tid], lsum[tid]);
        atomicAdd(&gsq[tid],  lsq[tid]);
    }
}

// =====================================================================
// 6) BatchNorm (batch statistics) + exact-erf GELU, in place on d_out.
// =====================================================================
__global__ void bn_gelu(float* __restrict__ out,
                        const float* __restrict__ gsum,
                        const float* __restrict__ gsq,
                        const float* __restrict__ gamma,
                        const float* __restrict__ beta) {
    long i = (long)blockIdx.x * 256 + threadIdx.x;
    int o = (int)((i >> 15) & 255);          // T*F = 32768 = 2^15
    const float inv_cnt = 1.0f / (float)MROW;
    float mean = gsum[o] * inv_cnt;
    float var  = gsq[o] * inv_cnt - mean * mean;
    float v = (out[i] - mean) * rsqrtf(var + 1e-5f);
    v = v * gamma[o] + beta[o];
    out[i] = 0.5f * v * (1.0f + erff(v * 0.70710678118654752f));
}

// =====================================================================
// launch
// =====================================================================
extern "C" void kernel_launch(void* const* d_in, const int* in_sizes, int n_in,
                              void* d_out, int out_size, void* d_ws, size_t ws_size,
                              hipStream_t stream) {
    const float* input = (const float*)d_in[0];
    const float* skip  = (const float*)d_in[1];
    const float* W_ih  = (const float*)d_in[2];
    const float* W_hh  = (const float*)d_in[3];
    const float* b_ih  = (const float*)d_in[4];
    const float* b_hh  = (const float*)d_in[5];
    const float* fc_w  = (const float*)d_in[6];
    const float* fc_b  = (const float*)d_in[7];
    const float* bn_g  = (const float*)d_in[8];
    const float* bn_b  = (const float*)d_in[9];
    float* out = (float*)d_out;
    char*  ws  = (char*)d_ws;

    // workspace layout (bytes)
    unsigned short* wih_t = (unsigned short*)(ws + 0);           // 48*8*512*2 = 384 KB
    unsigned short* whh_t = (unsigned short*)(ws + 393216);      // 384 KB
    unsigned short* fc_t  = (unsigned short*)(ws + 786432);      // 128 KB
    float*          gsum  = (float*)(ws + 917504);               // 1 KB
    float*          gsq   = gsum + 256;                          // 1 KB
    unsigned short* xb    = (unsigned short*)(ws + (1ul << 20));            // 64 MB
    unsigned short* gx    = (unsigned short*)(ws + (1ul << 20) + 67108864ul);   // 192 MB
    unsigned short* yg    = (unsigned short*)(ws + (1ul << 20) + 268435456ul);  // 64 MB

    // 1) repack weights into WMMA B-tile layout (bf16)
    pack_weights<<<768, 256, 0, stream>>>(W_ih, wih_t, 8, CC, 48 * 8 * 512);
    pack_weights<<<768, 256, 0, stream>>>(W_hh, whh_t, 8, HH, 48 * 8 * 512);
    pack_weights<<<256, 256, 0, stream>>>(fc_w, fc_t,  8, HH, 16 * 8 * 512);

    // 2) transpose input to [n,t,c] bf16
    transpose_in<<<dim3(CC / 32, FF / 32, BB * TT), dim3(32, 8), 0, stream>>>(input, xb);

    // 3) big input-projection GEMM
    gx_gemm<<<MROW / 16, 256, 0, stream>>>(xb, wih_t, b_ih, gx);

    // 4) zero BN statistic accumulators (capture-legal memset node)
    hipMemsetAsync(gsum, 0, 2 * OO * sizeof(float), stream);

    // 5) sequential GRU scan (16 persistent blocks, state in LDS)
    gru_scan<<<NSEQ / 16, 256, 0, stream>>>(gx, whh_t, b_hh, yg);

    // 6) 1x1 conv + skip + BN statistics
    conv_skip<<<MROW / 16, 256, 0, stream>>>(yg, fc_t, fc_b, skip, out, gsum, gsq);

    // 7) BatchNorm + exact GELU
    bn_gelu<<<(BB * OO * TT * FF) / 256, 256, 0, stream>>>(out, gsum, gsq, bn_g, bn_b);
}
